// HybridGCRNGATCell_83846351552829
// MI455X (gfx1250) — matrix operand order repaired and checked by reading, not compiled, not run on hardware
//
#include <hip/hip_runtime.h>
#include <math.h>

// ---------------------------------------------------------------------------
// MI455X (gfx1250) hybrid GCRN+GAT LSTM cell.
//  * all matmuls on v_wmma_f32_16x16x32_bf16 (wave32 WMMA, f32 accumulate)
//  * LDS tiles staged pre-formatted as packed bf16 in WMMA operand layout:
//    consumption is 2x ds_load_b128 per operand, conversions happen once.
//  * GAT attention streams Wh chunks with double-buffered
//    global_load_async_to_lds_b128 (ASYNCcnt), overlapping DMA with the
//    masked-softmax + WMMA work.
// Problem is compute-bound: ~11.6 GFLOP vs ~85 MB of L2-resident traffic.
// ---------------------------------------------------------------------------

typedef __attribute__((ext_vector_type(16))) __bf16 v16bf;
typedef __attribute__((ext_vector_type(8)))  float  v8f;

#define B_     32
#define N_     400
#define F_     64
#define H_     64
#define HEADS_ 4
#define DH_    64
#define C4H_   256
#define NT_    25   // N_/16 row tiles

#define USE_ASYNC_LDS 1

__device__ __forceinline__ int laneid() { return (int)(threadIdx.x & 31u); }

__device__ __forceinline__ v8f wmma_bf16(v16bf a, v16bf b, v8f c) {
  // D = A(16x32) * B(32x16) + C, bf16 in / f32 accumulate
  return __builtin_amdgcn_wmma_f32_16x16x32_bf16(false, a, false, b, (short)0, c,
                                                 false, false);
}

// ---------------------------------------------------------------------------
// Operand-formatted LDS staging (contiguous k order, same in A and B), bf16-packed.
// A-operand consumer: lane l holds row r=l&15; element e -> K=(l<16?0:16)+e.
// Inverse (producer owns tile column k = lane): sel=lane>=16,
// e = lane&15; element (row i, k) -> slot (i+16*sel, e).
// ---------------------------------------------------------------------------
__device__ __forceinline__ void stage_a_bf16(const float* __restrict__ src, int nrows,
                                             int ncols, int ld, int r0, int k0,
                                             v16bf* __restrict__ buf) {
  const int lane = laneid();
  const int sel  = (lane >= 16) ? 1 : 0;
  const int e    = lane & 15;
  __bf16* p = (__bf16*)buf;
  #pragma unroll
  for (int i = 0; i < 16; ++i) {
    const int r = r0 + i, c = k0 + lane;
    const float v = (r < nrows && c < ncols) ? src[(size_t)r * ld + c] : 0.f;
    p[(i + 16 * sel) * 16 + e] = (__bf16)v;
  }
}

// B-operand consumer: lane l holds col c=l&15; element e -> K=(l<16?0:16)+e.
// Inverse: element (k=rr, col=cc) -> slot (cc + 16*(rr>=16), rr&15).
__device__ __forceinline__ void stage_b_bf16(const float* __restrict__ src, int nrows,
                                             int ncols, int ld, int k0, int c0,
                                             v16bf* __restrict__ buf) {
  const int lane = laneid();
  __bf16* p = (__bf16*)buf;
  #pragma unroll
  for (int i = 0; i < 16; ++i) {
    const int eidx = i * 32 + lane;
    const int rr = eidx >> 4, cc = eidx & 15;
    const int r = k0 + rr, c = c0 + cc;
    const float v = (r < nrows && c < ncols) ? src[(size_t)r * ld + c] : 0.f;
    p[(cc + 16 * (rr >> 4)) * 16 + (rr & 15)] = (__bf16)v;
  }
}

// f32-in-LDS B gather (used by the attention kernel on async-DMA'd chunks).
__device__ __forceinline__ v16bf gather_b_f32(const float* __restrict__ lds, int ld, int c0) {
  const int lane = laneid();
  const int c  = c0 + (lane & 15);
  const int kb = (lane < 16) ? 0 : 16;
  v16bf b;
  #pragma unroll
  for (int e = 0; e < 16; ++e) b[e] = (__bf16)lds[(kb + e) * ld + c];
  return b;
}

#if USE_ASYNC_LDS
// low 32 bits of a generic pointer to a __shared__ object == LDS byte offset
__device__ __forceinline__ unsigned lds_off_u32(const void* p) {
  return (unsigned)(size_t)p;
}
__device__ __forceinline__ void async_ld_b128(unsigned ldsoff, const float* g) {
  asm volatile("global_load_async_to_lds_b128 %0, %1, off"
               :: "v"(ldsoff), "v"(g) : "memory");
}
__device__ __forceinline__ void s_wait_async0()  { asm volatile("s_wait_asynccnt 0x0"  ::: "memory"); }
__device__ __forceinline__ void s_wait_async8()  { asm volatile("s_wait_asynccnt 0x8"  ::: "memory"); }
__device__ __forceinline__ void s_wait_async16() { asm volatile("s_wait_asynccnt 0x10" ::: "memory"); }
// Issue nIssues b128 async loads per lane (nIssues*512 contiguous bytes).
__device__ __forceinline__ void issue_chunk(const float* gbase, unsigned ldsbase, int nIssues) {
  const int lane = laneid();
  for (int i = 0; i < nIssues; ++i) {
    const unsigned off = (unsigned)(i * 512 + lane * 16);
    async_ld_b128(ldsbase + off, gbase + (off >> 2));
  }
}
#endif

// ---------------------------------------------------------------------------
// K1: Wh[b,h,n,d] = sum_f src[b,n,f] * W[h,f,d]   (both branches)
// ---------------------------------------------------------------------------
__global__ void wh_kernel(const float* __restrict__ x, const float* __restrict__ h,
                          const float* __restrict__ Wx, const float* __restrict__ Wht,
                          float* __restrict__ WhX, float* __restrict__ WhH) {
  __shared__ v16bf laop[32];
  __shared__ v16bf lbop[32];
  int idx = blockIdx.x;
  const int ct = idx & 3;    idx >>= 2;
  const int rt = idx % NT_;  idx /= NT_;
  const int hh = idx & 3;    idx >>= 2;
  const int b  = idx & 31;   idx >>= 5;
  const int br = idx;  // 0 = x branch, 1 = h branch

  const float* src = (br ? h : x) + (size_t)b * N_ * F_;
  const float* W   = (br ? Wht : Wx) + hh * 64 * 64;
  float* dst = (br ? WhH : WhX) + (size_t)(b * HEADS_ + hh) * N_ * DH_;
  const int lane = laneid();

  v8f acc = {};
  for (int k0 = 0; k0 < F_; k0 += 32) {
    stage_a_bf16(src, N_, F_, F_, rt * 16, k0, laop);
    stage_b_bf16(W, 64, 64, 64, k0, ct * 16, lbop);
    __syncthreads();
    acc = wmma_bf16(laop[lane], lbop[lane], acc);
    __syncthreads();
  }
  const int roff = (lane < 16) ? 0 : 8;
  const int col  = ct * 16 + (lane & 15);
  #pragma unroll
  for (int j = 0; j < 8; ++j)
    dst[(size_t)(rt * 16 + j + roff) * DH_ + col] = acc[j];
}

// ---------------------------------------------------------------------------
// K2: AX[b] = A @ x[b], AH[b] = A @ h[b]    (M=400, K=400, N=64)
// ---------------------------------------------------------------------------
__global__ void ax_kernel(const float* __restrict__ A, const float* __restrict__ x,
                          const float* __restrict__ h, float* __restrict__ AX,
                          float* __restrict__ AH) {
  __shared__ v16bf laop[32];
  __shared__ v16bf lbop[32];
  int idx = blockIdx.x;
  const int ct = idx & 3;    idx >>= 2;
  const int rt = idx % NT_;  idx /= NT_;
  const int b  = idx & 31;   idx >>= 5;
  const int br = idx;

  const float* src = (br ? h : x) + (size_t)b * N_ * F_;
  float* dst = (br ? AH : AX) + (size_t)b * N_ * F_;
  const int lane = laneid();

  v8f acc = {};
  for (int k0 = 0; k0 < N_; k0 += 32) {
    if (k0 + 32 < N_)  // CDNA5 global_prefetch_b8 on the streamed adjacency row
      __builtin_prefetch(&A[(size_t)(rt * 16) * N_ + k0 + 32], 0, 0);
    stage_a_bf16(A, N_, N_, N_, rt * 16, k0, laop);
    stage_b_bf16(src, N_, F_, F_, k0, ct * 16, lbop);
    __syncthreads();
    acc = wmma_bf16(laop[lane], lbop[lane], acc);
    __syncthreads();
  }
  const int roff = (lane < 16) ? 0 : 8;
  const int col  = ct * 16 + (lane & 15);
  #pragma unroll
  for (int j = 0; j < 8; ++j)
    dst[(size_t)(rt * 16 + j + roff) * F_ + col] = acc[j];
}

// ---------------------------------------------------------------------------
// K3: gates_gcrn = AX@W_gcx + AH@W_gch + (b_gcx + b_gch)   (M=400,K=64,N=256)
// ---------------------------------------------------------------------------
__global__ void gcrn_kernel(const float* __restrict__ AX, const float* __restrict__ AH,
                            const float* __restrict__ Wgcx, const float* __restrict__ Wgch,
                            const float* __restrict__ bgcx, const float* __restrict__ bgch,
                            float* __restrict__ out) {
  __shared__ v16bf laop[32];
  __shared__ v16bf lbop[32];
  int idx = blockIdx.x;
  const int ct = idx & 15;   idx >>= 4;
  const int rt = idx % NT_;  idx /= NT_;
  const int b  = idx;
  const int lane = laneid();

  v8f acc = {};
  const float* srcs[2] = { AX + (size_t)b * N_ * F_, AH + (size_t)b * N_ * F_ };
  const float* Ws[2]   = { Wgcx, Wgch };
  for (int s = 0; s < 2; ++s) {
    for (int k0 = 0; k0 < F_; k0 += 32) {
      stage_a_bf16(srcs[s], N_, F_, F_, rt * 16, k0, laop);
      stage_b_bf16(Ws[s], 64, C4H_, C4H_, k0, ct * 16, lbop);
      __syncthreads();
      acc = wmma_bf16(laop[lane], lbop[lane], acc);
      __syncthreads();
    }
  }
  const int roff = (lane < 16) ? 0 : 8;
  const int col  = ct * 16 + (lane & 15);
  const float bias = bgcx[col] + bgch[col];
  #pragma unroll
  for (int j = 0; j < 8; ++j)
    out[((size_t)b * N_ + rt * 16 + j + roff) * C4H_ + col] = acc[j] + bias;
}

// ---------------------------------------------------------------------------
// K4: es/ed = Wh . a_src / a_dst  (row reductions over d=64, both branches)
// ---------------------------------------------------------------------------
__global__ void esed_kernel(const float* __restrict__ WhX, const float* __restrict__ WhH,
                            const float* __restrict__ asx, const float* __restrict__ adx,
                            const float* __restrict__ ash, const float* __restrict__ adh,
                            float* __restrict__ esx, float* __restrict__ edx,
                            float* __restrict__ esh, float* __restrict__ edh) {
  const int total = 2 * B_ * HEADS_ * N_;
  int t = blockIdx.x * blockDim.x + threadIdx.x;
  if (t >= total) return;
  const int br = t / (B_ * HEADS_ * N_);
  const int r  = t % (B_ * HEADS_ * N_);
  const int hh = (r / N_) % HEADS_;
  const float* Wh = (br ? WhH : WhX) + (size_t)r * DH_;
  const float* as = (br ? ash : asx) + hh * DH_;
  const float* ad = (br ? adh : adx) + hh * DH_;
  float s = 0.f, d = 0.f;
  #pragma unroll 8
  for (int k = 0; k < DH_; ++k) { const float w = Wh[k]; s += w * as[k]; d += w * ad[k]; }
  (br ? esh : esx)[r] = s;
  (br ? edh : edx)[r] = d;
}

// ---------------------------------------------------------------------------
// K5: masked-softmax GAT attention, flash-style, both branches summed.
// One wave per (b, head, 16-row tile).  P tile is written straight into WMMA
// A-operand format; Wh chunks are DMA'd by double-buffered async loads so the
// next chunk streams in while the current chunk's exp/WMMA work runs.
// Normalization by row-sum at the end (masked entries are exact zeros; logits
// are O(1) so no max-subtraction is needed; empty rows guarded).
// ---------------------------------------------------------------------------
__global__ void gat_attn_kernel(const float* __restrict__ A,
                                const float* __restrict__ WhX, const float* __restrict__ WhH,
                                const float* __restrict__ esx, const float* __restrict__ edx,
                                const float* __restrict__ esh, const float* __restrict__ edh,
                                float* __restrict__ gat) {
  __shared__ float les[16];
  __shared__ v16bf lpa[32];          // P tile, WMMA A-operand formatted
  __shared__ float lwh[2][32 * 64];  // double-buffered raw f32 Wh chunk
  __shared__ float lrs[16];
  int idx = blockIdx.x;
  const int rt = idx % NT_;  idx /= NT_;
  const int hh = idx & 3;    idx >>= 2;
  const int b  = idx;
  const int lane = laneid();
  const size_t bh = (size_t)(b * HEADS_ + hh);
  const int n0 = rt * 16;

  const int psel = (lane >= 16) ? 1 : 0;                   // operand slot
  const int pe   = lane & 15;  // for column `lane`
  __bf16* lpae = (__bf16*)lpa;

  v8f res[4] = {};
  for (int br = 0; br < 2; ++br) {
    const float* Wh = (br ? WhH : WhX) + bh * N_ * DH_;
    const float* es = (br ? esh : esx) + bh * N_;
    const float* ed = (br ? edh : edx) + bh * N_;
    v8f acc[4] = {};
    if (lane < 16) les[lane] = es[n0 + lane];
    float rs = 0.f;
#if USE_ASYNC_LDS
    issue_chunk(Wh, lds_off_u32(&lwh[0][0]), 16);   // prologue: chunk 0 (32 rows)
#endif
    __syncthreads();
    int buf = 0;
    for (int m0 = 0; m0 < N_; m0 += 32) {
      const int nxt = m0 + 32;
#if USE_ASYNC_LDS
      int nxt_issues = 0;
      if (nxt < N_) {                           // prefetch next chunk via DMA
        const int nrows = (nxt + 32 <= N_) ? 32 : (N_ - nxt);  // 32 or 16
        nxt_issues = nrows >> 1;                // b128 issues per lane
        issue_chunk(Wh + (size_t)nxt * DH_, lds_off_u32(&lwh[buf ^ 1][0]), nxt_issues);
        if (nrows == 16)                        // zero-pad rows 16..31
          for (int i = 32; i < 64; ++i) lwh[buf ^ 1][i * 32 + lane] = 0.f;
      }
#else
      {
        const int nrows = (m0 + 32 <= N_) ? 32 : (N_ - m0);
        for (int i = 0; i < 64; ++i) {
          const int e = i * 32 + lane;
          const int rr = e >> 6, cc = e & 63;
          lwh[buf][e] = (rr < nrows) ? Wh[(size_t)(m0 + rr) * DH_ + cc] : 0.f;
        }
      }
#endif
      // ---- P tile: masked exp(leaky_relu(es+ed)), stored pre-formatted ----
      const int  mcol  = m0 + lane;
      const bool colok = mcol < N_;
      const float edv  = colok ? ed[mcol] : 0.f;
      #pragma unroll
      for (int i = 0; i < 16; ++i) {
        float p = 0.f;
        if (colok) {
          const float av = A[(size_t)(n0 + i) * N_ + mcol];
          if (av > 1e-6f) {
            float e = les[i] + edv;
            e = (e > 0.f) ? e : 0.2f * e;       // leaky_relu(0.2)
            p = __expf(e);
          }
        }
        lpae[(i + 16 * psel) * 16 + pe] = (__bf16)p;
      }
      __syncthreads();
#if USE_ASYNC_LDS
      if (nxt_issues == 16)     s_wait_async16();  // current chunk landed,
      else if (nxt_issues == 8) s_wait_async8();   // next still in flight
      else                      s_wait_async0();
#endif
      const v16bf pa = lpa[lane];
      {  // chunk row-sums from the operand register + cross-half shuffle
        float s = 0.f;
        #pragma unroll
        for (int e = 0; e < 16; ++e) s += (float)pa[e];
        s += __shfl_xor(s, 16, 32);
        rs += s;                 // every lane holds the sum of row (lane & 15)
      }
      #pragma unroll
      for (int ct = 0; ct < 4; ++ct)
        acc[ct] = wmma_bf16(pa, gather_b_f32(&lwh[buf][0], 64, ct * 16), acc[ct]);
      __syncthreads();
      buf ^= 1;
    }
    if (lane < 16) lrs[lane] = rs;
    __syncthreads();
    const int roff = (lane < 16) ? 0 : 8;
    #pragma unroll
    for (int j = 0; j < 8; ++j) {
      const float rsum = lrs[j + roff];
      const float rinv = (rsum > 0.f) ? (1.f / rsum) : 0.f;
      #pragma unroll
      for (int ct = 0; ct < 4; ++ct) res[ct][j] += acc[ct][j] * rinv;
    }
    __syncthreads();
  }
  const int roff = (lane < 16) ? 0 : 8;
  #pragma unroll
  for (int j = 0; j < 8; ++j) {
    const int n = n0 + j + roff;
    #pragma unroll
    for (int ct = 0; ct < 4; ++ct)
      gat[((size_t)b * N_ + n) * C4H_ + hh * DH_ + ct * 16 + (lane & 15)] = res[ct][j];
  }
}

// ---------------------------------------------------------------------------
// K6: fused = [gcrn | gat] @ W_fp + b_fp ; LSTM gates ; h_t, c_t.
// One wave owns the i/f/o/g tiles of a (b, row-tile, 16-col slab): they share
// the K-loop/A operand and the nonlinearities + cell update stay in-register.
// ---------------------------------------------------------------------------
__global__ void fuse_lstm_kernel(const float* __restrict__ gcrn, const float* __restrict__ gat,
                                 const float* __restrict__ Wfp, const float* __restrict__ bfp,
                                 const float* __restrict__ cprev,
                                 float* __restrict__ ht, float* __restrict__ ct_out) {
  __shared__ v16bf laop[32];
  __shared__ v16bf lbop4[4][32];
  int idx = blockIdx.x;
  const int ct = idx & 3;    idx >>= 2;
  const int rt = idx % NT_;  idx /= NT_;
  const int b  = idx;
  const int lane = laneid();
  __bf16* lbe = (__bf16*)lbop4;

  v8f acc[4] = {};  // i, f, o, g tiles
  for (int kc = 0; kc < 16; ++kc) {   // K = 512 in chunks of 32
    const float* src = (kc < 8) ? gcrn : gat;
    const int k0 = (kc & 7) * 32;
    stage_a_bf16(src + (size_t)b * N_ * C4H_, N_, C4H_, C4H_, rt * 16, k0, laop);
    for (int i = 0; i < 64; ++i) {    // stage 4 gate B-tiles, operand formatted
      const int eidx = i * 32 + lane;
      const int rr = eidx >> 6, cc = eidx & 63;   // k row 0..31, packed col 0..63
      const int g = cc >> 4, c16 = cc & 15;
      const float v = Wfp[(size_t)(kc * 32 + rr) * C4H_ + g * 64 + ct * 16 + c16];
      lbe[(g * 32 + c16 + 16 * (rr >> 4)) * 16 + (rr & 15)] = (__bf16)v;
    }
    __syncthreads();
    const v16bf av = laop[lane];
    #pragma unroll
    for (int g = 0; g < 4; ++g)
      acc[g] = wmma_bf16(av, lbop4[g][lane], acc[g]);
    __syncthreads();
  }
  const int hc = ct * 16 + (lane & 15);
  const float bi = bfp[hc], bf = bfp[64 + hc], bo = bfp[128 + hc], bg = bfp[192 + hc];
  const int roff = (lane < 16) ? 0 : 8;
  #pragma unroll
  for (int j = 0; j < 8; ++j) {
    const int n = rt * 16 + j + roff;
    const size_t p = ((size_t)b * N_ + n) * H_ + hc;
    const float iv = 1.f / (1.f + __expf(-(acc[0][j] + bi)));
    const float fv = 1.f / (1.f + __expf(-(acc[1][j] + bf)));
    const float ov = 1.f / (1.f + __expf(-(acc[2][j] + bo)));
    const float gv = tanhf(acc[3][j] + bg);
    const float cv = fv * cprev[p] + iv * gv;
    ht[p]     = ov * tanhf(cv);
    ct_out[p] = cv;
  }
}

// ---------------------------------------------------------------------------
extern "C" void kernel_launch(void* const* d_in, const int* in_sizes, int n_in,
                              void* d_out, int out_size, void* d_ws, size_t ws_size,
                              hipStream_t stream) {
  (void)in_sizes; (void)n_in; (void)out_size; (void)ws_size;
  const float* x_t    = (const float*)d_in[0];
  const float* h_prev = (const float*)d_in[1];
  const float* c_prev = (const float*)d_in[2];
  const float* A      = (const float*)d_in[3];
  const float* Wgcx   = (const float*)d_in[4];
  const float* bgcx   = (const float*)d_in[5];
  const float* Wgch   = (const float*)d_in[6];
  const float* bgch   = (const float*)d_in[7];
  const float* Wgatx  = (const float*)d_in[8];
  const float* asx    = (const float*)d_in[9];
  const float* adx    = (const float*)d_in[10];
  const float* Wgath  = (const float*)d_in[11];
  const float* ash    = (const float*)d_in[12];
  const float* adh    = (const float*)d_in[13];
  const float* Wfp    = (const float*)d_in[14];
  const float* bfp    = (const float*)d_in[15];

  // Workspace layout (floats). Total ~14.95 M floats (~59.8 MB) — L2-resident.
  float* ws   = (float*)d_ws;
  float* AX   = ws;                 // [B,N,64]      819200
  float* AH   = AX   + 819200;      // [B,N,64]      819200
  float* WhX  = AH   + 819200;      // [B,4,N,64]    3276800
  float* WhH  = WhX  + 3276800;     // [B,4,N,64]    3276800
  float* esx  = WhH  + 3276800;     // [B,4,N]       51200
  float* edx  = esx  + 51200;
  float* esh  = edx  + 51200;
  float* edh  = esh  + 51200;
  float* gcrn = edh  + 51200;       // [B,N,256]     3276800
  float* gat  = gcrn + 3276800;     // [B,N,256]     3276800

  float* ht = (float*)d_out;
  float* ct = ht + (size_t)B_ * N_ * H_;

  wh_kernel       <<<2 * B_ * HEADS_ * NT_ * 4, 32, 0, stream>>>(x_t, h_prev, Wgatx, Wgath, WhX, WhH);
  ax_kernel       <<<2 * B_ * NT_ * 4,          32, 0, stream>>>(A, x_t, h_prev, AX, AH);
  gcrn_kernel     <<<B_ * NT_ * 16,             32, 0, stream>>>(AX, AH, Wgcx, Wgch, bgcx, bgch, gcrn);
  esed_kernel     <<<(2 * B_ * HEADS_ * N_ + 255) / 256, 256, 0, stream>>>(WhX, WhH, asx, adx, ash, adh,
                                                                           esx, edx, esh, edh);
  gat_attn_kernel <<<B_ * HEADS_ * NT_,         32, 0, stream>>>(A, WhX, WhH, esx, edx, esh, edh, gat);
  fuse_lstm_kernel<<<B_ * NT_ * 4,              32, 0, stream>>>(gcrn, gat, Wfp, bfp, c_prev, ht, ct);
}
